// QConv_49014166781968
// MI455X (gfx1250) — matrix operand order repaired and checked by reading, not compiled, not run on hardware
//
#include <hip/hip_runtime.h>
#include <cstdint>
#include <cstddef>

#define N_NODES 100000
#define N_EDGES 1600000
#define FEAT 128
#define K1   131      // in_feat + 3 edge feats
#define K1PAD 160     // 5 x 32 (chunk 4 = edge feats + zero pad)
#define K2   256      // in_feat + inter

#define EDGE_BLOCKS 2048
#define NODE_BLOCKS 512
#define ZERO_BLOCKS 2048

typedef __attribute__((ext_vector_type(16))) __bf16 v16bf;
typedef __attribute__((ext_vector_type(8)))  __bf16 v8bf;
typedef __attribute__((ext_vector_type(8)))  float  v8f;

__device__ __forceinline__ v8bf cvt8(float4 a, float4 b) {
    v8bf r;
    r[0] = (__bf16)a.x; r[1] = (__bf16)a.y; r[2] = (__bf16)a.z; r[3] = (__bf16)a.w;
    r[4] = (__bf16)b.x; r[5] = (__bf16)b.y; r[6] = (__bf16)b.z; r[7] = (__bf16)b.w;
    return r;
}
__device__ __forceinline__ v16bf cat16(v8bf lo, v8bf hi) {
    return __builtin_shufflevector(lo, hi, 0,1,2,3,4,5,6,7,8,9,10,11,12,13,14,15);
}
// Build one A-fragment (16x32 bf16, this lane's share) straight from a global
// f32 row. Layout: lane holds row (lane&15); K = k0..k0+15
// where k0 = kcBase + (lane>>4)*16.
__device__ __forceinline__ v16bf load_a_frag(const float* __restrict__ row, int kcBase, int half) {
    const float* p = row + kcBase + half * 16;
    return cat16(cvt8(*(const float4*)(p),      *(const float4*)(p + 4)),
                 cvt8(*(const float4*)(p + 8), *(const float4*)(p + 12)));
}

__global__ __launch_bounds__(256) void zero_f32(float* __restrict__ p, int n4) {
    const int stride = gridDim.x * blockDim.x;
    for (int i = blockIdx.x * blockDim.x + threadIdx.x; i < n4; i += stride)
        ((float4*)p)[i] = make_float4(0.f, 0.f, 0.f, 0.f);
}

// ---------------------------------------------------------------------------
// Kernel 1: per-edge MLP (concat(h[src],w) @ W1^T + b1, relu) + scatter-add.
// One wave = one 16-edge x 128-out tile; K split into 5 chunks of 32.
// ---------------------------------------------------------------------------
__global__ __launch_bounds__(256) void edge_mlp_scatter(
    const float* __restrict__ h, const float* __restrict__ ew,
    const int* __restrict__ src, const int* __restrict__ dst,
    const float* __restrict__ W1, const float* __restrict__ b1,
    float* __restrict__ hN, int nTiles, int waveStride)
{
    // W1 as bf16, rows = output channel, padded K=160, zero-filled pad. 40 KB.
    alignas(32) __shared__ __bf16 sW1[FEAT * K1PAD];
    for (int i = threadIdx.x; i < FEAT * K1PAD; i += 256) {
        const int r = i / K1PAD, c = i - r * K1PAD;
        sW1[i] = (__bf16)((c < K1) ? W1[r * K1 + c] : 0.0f);
    }
    __syncthreads();

    const int lane  = threadIdx.x & 31;
    const int wave  = threadIdx.x >> 5;
    const int row16 = lane & 15;
    const int half  = lane >> 4;

    for (int tile = blockIdx.x * 8 + wave; tile < nTiles; tile += waveStride) {
        const int e0   = tile * 16;
        const int eRow = e0 + row16;
        const int s    = __builtin_nontemporal_load(src + eRow);
        const float* __restrict__ hrow = h + (size_t)s * FEAT;

        // Prefetch next tile's gathered row into cache while we compute.
        const int tNext = tile + waveStride;
        if (tNext < nTiles) {
            const int sn = __builtin_nontemporal_load(src + tNext * 16 + row16);
            __builtin_prefetch(h + (size_t)sn * FEAT, 0, 1);
        }

        // A fragments: chunks 0..3 from h[src], chunk 4 holds the 3 edge feats.
        v16bf a[5];
#pragma unroll
        for (int kc = 0; kc < 4; ++kc)
            a[kc] = load_a_frag(hrow, kc * 32, half);
        {
            v16bf t = {};
            if (half == 0) {   // K=128..135 live in half 0, elements 0..2 valid
                const float* wp = ew + (size_t)eRow * 3;
                t[0] = (__bf16)wp[0]; t[1] = (__bf16)wp[1]; t[2] = (__bf16)wp[2];
            }
            a[4] = t;
        }

        const int rbase = half * 8;   // D rows owned by this lane
        int dIdx[8];
#pragma unroll
        for (int r = 0; r < 8; ++r)
            dIdx[r] = __builtin_nontemporal_load(dst + e0 + rbase + r);

#pragma unroll
        for (int nc = 0; nc < 8; ++nc) {
            v8f c = {};
            // B fragment: lane holds col n = row16 of W1 chunk, 16 consecutive K
            const char* wp = (const char*)sW1 +
                             (((nc * 16 + row16) * K1PAD + half * 16) * 2);
#pragma unroll
            for (int kc = 0; kc < 5; ++kc) {
                const v16bf bfr = *(const v16bf*)(wp + kc * 64);
                c = __builtin_amdgcn_wmma_f32_16x16x32_bf16(
                        false, a[kc], false, bfr, (short)0, c, false, false);
            }
            const float bias = b1[nc * 16 + row16];
            float* outBase = hN + nc * 16 + row16;
#pragma unroll
            for (int r = 0; r < 8; ++r) {
                const float v = fmaxf(c[r] + bias, 0.0f);
                unsafeAtomicAdd(outBase + (size_t)dIdx[r] * FEAT, v);
            }
        }
    }
}

// ---------------------------------------------------------------------------
// Kernel 2: per-node MLP: relu(concat(h, h_N) @ W2^T). K=256 -> 8 chunks.
// ---------------------------------------------------------------------------
__global__ __launch_bounds__(256) void node_mlp(
    const float* __restrict__ h, const float* __restrict__ hN,
    const float* __restrict__ W2, float* __restrict__ out,
    int nTiles, int waveStride)
{
    // W2 as bf16, 64 KB, XOR-swizzled (64B granules) to break 512B-stride banks
    alignas(64) __shared__ __bf16 sW2[FEAT * K2];
    for (int i = threadIdx.x; i < FEAT * K2; i += 256) {
        const int r = i >> 8, k = i & 255;
        const int boff = (r << 9) | ((k << 1) ^ ((r & 7) << 6));
        *(__bf16*)((char*)sW2 + boff) = (__bf16)W2[i];
    }
    __syncthreads();

    const int lane  = threadIdx.x & 31;
    const int wave  = threadIdx.x >> 5;
    const int row16 = lane & 15;
    const int half  = lane >> 4;

    for (int tile = blockIdx.x * 8 + wave; tile < nTiles; tile += waveStride) {
        const int n0 = tile * 16;
        const float* __restrict__ hrow = h  + (size_t)(n0 + row16) * FEAT;
        const float* __restrict__ grow = hN + (size_t)(n0 + row16) * FEAT;

        v16bf a[8];
#pragma unroll
        for (int kc = 0; kc < 4; ++kc) {
            a[kc]     = load_a_frag(hrow, kc * 32, half);   // K 0..127  = h
            a[kc + 4] = load_a_frag(grow, kc * 32, half);   // K 128..255 = h_N
        }

        const int rbase = half * 8;
#pragma unroll
        for (int nc = 0; nc < 8; ++nc) {
            v8f c = {};
            const int wrow = nc * 16 + row16;
#pragma unroll
            for (int kc = 0; kc < 8; ++kc) {
                const int boff = (wrow << 9) |
                                 ((kc * 64 + half * 32) ^ ((wrow & 7) << 6));
                const v16bf bfr = *(const v16bf*)((const char*)sW2 + boff);
                c = __builtin_amdgcn_wmma_f32_16x16x32_bf16(
                        false, a[kc], false, bfr, (short)0, c, false, false);
            }
            float* obase = out + nc * 16 + row16;
#pragma unroll
            for (int r = 0; r < 8; ++r)
                obase[(size_t)(n0 + rbase + r) * FEAT] = fmaxf(c[r], 0.0f);
        }
    }
}

extern "C" void kernel_launch(void* const* d_in, const int* in_sizes, int n_in,
                              void* d_out, int out_size, void* d_ws, size_t ws_size,
                              hipStream_t stream) {
    const float* h   = (const float*)d_in[0];
    const float* ew  = (const float*)d_in[1];
    const int*   src = (const int*)d_in[2];
    const int*   dst = (const int*)d_in[3];
    const float* W1  = (const float*)d_in[4];
    const float* b1  = (const float*)d_in[5];
    const float* W2  = (const float*)d_in[6];
    float* out = (float*)d_out;
    float* hN  = (float*)d_ws;                 // 100000*128 f32 = 51.2 MB scratch

    zero_f32<<<ZERO_BLOCKS, 256, 0, stream>>>(hN, (N_NODES * FEAT) / 4);

    const int eTiles = N_EDGES / 16;           // 100000, exact
    edge_mlp_scatter<<<EDGE_BLOCKS, 256, 0, stream>>>(
        h, ew, src, dst, W1, b1, hN, eTiles, EDGE_BLOCKS * 8);

    const int nTiles = N_NODES / 16;           // 6250, exact
    node_mlp<<<NODE_BLOCKS, 256, 0, stream>>>(
        h, hN, W2, out, nTiles, NODE_BLOCKS * 8);
}
